// TransformerEncoder_78486232367472
// MI455X (gfx1250) — matrix operand order repaired and checked by reading, not compiled, not run on hardware
//
#include <hip/hip_runtime.h>
#include <hip/hip_bf16.h>
#include <math.h>

// ---------------------------------------------------------------------------
// Problem constants (from the reference)
// ---------------------------------------------------------------------------
#define D_MODEL   768
#define N_HEADS   8
#define HEAD_DIM  96
#define FFN_DIM   3072
#define BATCH     8
#define NTOK      512
#define PREFIX_N  16
#define SEQ       1040                  // 16 + 2*512
#define ROWS      (BATCH * SEQ)         // 8320 = 65 * 128
#define MLP_HID_N 4096
#define OUT_DIM_N 1024
#define QKV_N     (3 * D_MODEL)         // 2304

typedef __attribute__((ext_vector_type(16))) __bf16          v16bf;
typedef __attribute__((ext_vector_type(16))) unsigned short  v16u;
typedef __attribute__((ext_vector_type(8)))  float           v8f;
typedef __attribute__((ext_vector_type(4)))  float           v4f;
typedef __attribute__((ext_vector_type(4)))  __bf16          v4bf;

// float4 -> packed bf16x4 (RNE) as one 64-bit word; native v_cvt_pk_bf16_f32
__device__ __forceinline__ unsigned long long f4_to_bf4(v4f x) {
  v4bf b = __builtin_convertvector(x, v4bf);
  return __builtin_bit_cast(unsigned long long, b);
}
__device__ __forceinline__ unsigned short f2bf(float f) {
  __bf16 b = (__bf16)f;
  return __builtin_bit_cast(unsigned short, b);
}

// Permutes K within a 32-wide chunk into WMMA A-fragment order:
// lane-half h reads contiguous [h*16, h*16+16): elems 0-7 <-> K h*8+0..7,
// elems 8-15 <-> K h*8+16..23.  slot(k) = (k&7) | ((k&8)<<1) | ((k&16)>>1)
__device__ __forceinline__ int kslot(int k) {
  return (k & 7) | ((k & 8) << 1) | ((k & 16) >> 1);
}

__device__ __forceinline__ v8f wmma_bf16(const v16u a, const v16u b, const v8f c) {
  // (neg_a, A, neg_b, B, c_mod, C, reuse_a, reuse_b)
  return __builtin_amdgcn_wmma_f32_16x16x32_bf16(
      false, __builtin_bit_cast(v16bf, a),
      false, __builtin_bit_cast(v16bf, b),
      (short)0, c, false, false);
}

// ---------------------------------------------------------------------------
// Generic GEMM: C[M,N] = act(A[M,K] @ W[K,N] + bias[N])
// fp32 in/out, bf16 WMMA with fp32 accumulation. Block = 8 wave32 (4Mx2N),
// block tile 128x64, K step 32, double-buffered LDS (one barrier per K step):
// tile t+1 is loaded from global while WMMAs consume tile t. LDS tiles are in
// fragment order so each WMMA operand is one contiguous 32B v16u load.
// M-masking uses clamped-index loads + select-zero (no exec branching).
// Requires N % 64 == 0 and K % 32 == 0 (true at all call sites).
// ---------------------------------------------------------------------------
__global__ __launch_bounds__(256) void gemm_bf16_kernel(
    const float* __restrict__ A, long lda,
    const float* __restrict__ W, const float* __restrict__ bias,
    float* __restrict__ C, long ldc, int M, int N, int K, int act) {
  __shared__ __align__(32) unsigned short As[2][128 * 32];  // [m][slot(k)]
  __shared__ __align__(32) unsigned short Bst[2][64 * 32];  // [n][k] transposed

  const int tid  = threadIdx.x;
  const int lane = tid & 31;
  const int wave = tid >> 5;
  const int wm   = wave & 3, wn = wave >> 2;
  const int ml   = lane & 15, half = lane >> 4;
  const int rowBlock = blockIdx.y * 128;
  const int colBlock = blockIdx.x * 64;
  const bool fullM = (rowBlock + 128 <= M);

  // Stage one 128x32 A tile + 32x64 W tile into LDS buffer `buf`.
  // Loads are issued first (6 b128 requests in flight), then convert+store.
  auto stage = [&](int buf, int k0) {
    v4f av[4], wv[2];
#pragma unroll
    for (int i = 0; i < 4; ++i) {
      int f = i * 256 + tid;               // 1024 float4 = 128x32
      int row = f >> 3, c4 = (f & 7) * 4;
      int grow = rowBlock + row;
      int rsafe = grow < (M - 1) ? grow : (M - 1);
      av[i] = *(const v4f*)&A[(long)rsafe * lda + k0 + c4];
    }
#pragma unroll
    for (int i = 0; i < 2; ++i) {
      int f = i * 256 + tid;               // 512 float4 = 32x64
      int kk = f >> 4, n4 = (f & 15) * 4;
      wv[i] = *(const v4f*)&W[(long)(k0 + kk) * N + colBlock + n4];
    }
#pragma unroll
    for (int i = 0; i < 4; ++i) {
      int f = i * 256 + tid;
      int row = f >> 3, c4 = (f & 7) * 4;
      unsigned long long pk = f4_to_bf4(av[i]);
      if (rowBlock + row >= M) pk = 0ull;
      *(unsigned long long*)&As[buf][row * 32 + kslot(c4)] = pk;
    }
#pragma unroll
    for (int i = 0; i < 2; ++i) {
      int f = i * 256 + tid;
      int kk = f >> 4, n4 = (f & 15) * 4;
      unsigned long long pk = f4_to_bf4(wv[i]);
      Bst[buf][(n4 + 0) * 32 + kslot(kk)] = (unsigned short)(pk);
      Bst[buf][(n4 + 1) * 32 + kslot(kk)] = (unsigned short)(pk >> 16);
      Bst[buf][(n4 + 2) * 32 + kslot(kk)] = (unsigned short)(pk >> 32);
      Bst[buf][(n4 + 3) * 32 + kslot(kk)] = (unsigned short)(pk >> 48);
    }
  };

  v8f acc[2][2] = {};
  const int nk = K >> 5;

  stage(0, 0);
  __syncthreads();
  for (int t = 0; t < nk; ++t) {
    const int cur = t & 1;
    if (t + 1 < nk) stage(cur ^ 1, (t + 1) * 32);  // overlap with WMMAs below

    v16u a[2], bf[2];
#pragma unroll
    for (int tm = 0; tm < 2; ++tm)
      a[tm] = *(const v16u*)&As[cur][(wm * 32 + tm * 16 + ml) * 32 + half * 16];
#pragma unroll
    for (int tn = 0; tn < 2; ++tn)
      bf[tn] = *(const v16u*)&Bst[cur][(wn * 32 + tn * 16 + ml) * 32 + half * 16];
#pragma unroll
    for (int tm = 0; tm < 2; ++tm)
#pragma unroll
      for (int tn = 0; tn < 2; ++tn)
        acc[tm][tn] = wmma_bf16(a[tm], bf[tn], acc[tm][tn]);
    __syncthreads();   // one barrier per K step
  }

  // Epilogue: C layout -> lanes 0-15 rows r, lanes 16-31 rows 8+r, col = lane&15
#pragma unroll
  for (int tm = 0; tm < 2; ++tm)
#pragma unroll
    for (int tn = 0; tn < 2; ++tn)
#pragma unroll
      for (int r = 0; r < 8; ++r) {
        int row = rowBlock + wm * 32 + tm * 16 + ((lane < 16) ? r : 8 + r);
        int col = colBlock + wn * 32 + tn * 16 + ml;
        if (fullM || row < M) {
          float vv = acc[tm][tn][r] + bias[col];
          if (act == 1) vv = 0.5f * vv * (1.0f + erff(vv * 0.70710678118f)); // exact GELU
          C[(long)row * ldc + col] = vv;
        }
      }
}

// ---------------------------------------------------------------------------
// Flash attention: one wave32 per (b, h, 16-row q tile). q/k/v are views into
// the fused qkv buffer with row stride ldq (2304). Online softmax; K/V chunks
// of 32 rows; scale folded into q. LDS layouts give contiguous v16u fragments:
//   qs[m][kc*32 + slot]   ks[krow][dh]   vst[dh][krow]   ps[m][slot]
// ---------------------------------------------------------------------------
__global__ __launch_bounds__(32) void attn_kernel(
    const float* __restrict__ q, const float* __restrict__ k,
    const float* __restrict__ v, float* __restrict__ out, long ldq) {
  const int bid = blockIdx.x;
  const int qt  = bid % (SEQ / 16);
  const int h   = (bid / (SEQ / 16)) % N_HEADS;
  const int b   = bid / ((SEQ / 16) * N_HEADS);
  const int lane = threadIdx.x;
  const int ml = lane & 15, half = lane >> 4;

  __shared__ __align__(32) unsigned short qs[16 * 96];
  __shared__ __align__(32) unsigned short ks[32 * 96];
  __shared__ __align__(32) unsigned short vst[96 * 32];
  __shared__ __align__(32) unsigned short ps[16 * 32];

  const float scale = 0.10206207262f;  // 1/sqrt(96)

  // ---- stage q tile (16x96), scaled, in A-fragment order ----
#pragma unroll
  for (int g = 0; g < 3; ++g) {
    v4f qv[4];
#pragma unroll
    for (int j = 0; j < 4; ++j) {
      int f = (g * 4 + j) * 32 + lane;   // 384 float4 = 16x96
      int m = f / 24, c4 = (f % 24) * 4;
      int srow = qt * 16 + m;
      qv[j] = *(const v4f*)&q[((long)(b * SEQ + srow)) * ldq + h * HEAD_DIM + c4];
    }
#pragma unroll
    for (int j = 0; j < 4; ++j) {
      int f = (g * 4 + j) * 32 + lane;
      int m = f / 24, c4 = (f % 24) * 4;
      v4f s4 = qv[j] * scale;
      int kc = c4 >> 5, wi = c4 & 31;
      *(unsigned long long*)&qs[m * 96 + kc * 32 + kslot(wi)] = f4_to_bf4(s4);
    }
  }
  __syncthreads();

  v16u qa[3];
#pragma unroll
  for (int kc = 0; kc < 3; ++kc)
    qa[kc] = *(const v16u*)&qs[ml * 96 + kc * 32 + half * 16];

  v8f o_acc[6] = {};
  float m_run[8], l_run[8];
#pragma unroll
  for (int r = 0; r < 8; ++r) { m_run[r] = -3.0e38f; l_run[r] = 0.f; }

  const int nkb = (SEQ + 31) / 32;       // 33 (last chunk half-valid)
  for (int kb = 0; kb < nkb; ++kb) {
    int krow = kb * 32 + lane;           // one k/v row per lane
    const bool valid = krow < SEQ;
    int ksafe = valid ? krow : (SEQ - 1);
    const float* kp = k + ((long)(b * SEQ + ksafe)) * ldq + h * HEAD_DIM;
    const float* vp = v + ((long)(b * SEQ + ksafe)) * ldq + h * HEAD_DIM;
    // groups of 4 float4 so 8 b128 loads are in flight before converts/stores
#pragma unroll
    for (int g = 0; g < 6; ++g) {
      v4f kv[4], vv[4];
#pragma unroll
      for (int j = 0; j < 4; ++j) {
        kv[j] = *(const v4f*)(kp + (g * 4 + j) * 4);
        vv[j] = *(const v4f*)(vp + (g * 4 + j) * 4);
      }
#pragma unroll
      for (int j = 0; j < 4; ++j) {
        int c4 = g * 4 + j;
        unsigned long long pk = f4_to_bf4(kv[j]);
        if (!valid) pk = 0ull;
        *(unsigned long long*)&ks[lane * 96 + ((c4 * 4) & ~31) + kslot((c4 * 4) & 31)] = pk;
        unsigned long long pv = f4_to_bf4(vv[j]);
        if (!valid) pv = 0ull;
        vst[(c4 * 4 + 0) * 32 + kslot(lane)] = (unsigned short)(pv);
        vst[(c4 * 4 + 1) * 32 + kslot(lane)] = (unsigned short)(pv >> 16);
        vst[(c4 * 4 + 2) * 32 + kslot(lane)] = (unsigned short)(pv >> 32);
        vst[(c4 * 4 + 3) * 32 + kslot(lane)] = (unsigned short)(pv >> 48);
      }
    }
    __syncthreads();

    // scores s[16 x 32] = q @ k^T (two 16x16 tiles, K=dh=96 -> 3 WMMAs each)
    v8f s[2] = {};
#pragma unroll
    for (int t = 0; t < 2; ++t) {
#pragma unroll
      for (int kc = 0; kc < 3; ++kc) {
        v16u bb = *(const v16u*)&ks[(t * 16 + ml) * 96 + kc * 32 + half * 16];
        s[t] = wmma_bf16(qa[kc], bb, s[t]);
      }
      int col = kb * 32 + t * 16 + ml;
      if (col >= SEQ)
#pragma unroll
        for (int r = 0; r < 8; ++r) s[t][r] = -3.0e38f;
    }

    // Online softmax. C layout: this lane owns rows r (+8 for upper half),
    // one column each; row reduce = shfl-xor across the 16-lane group.
    float alpha[8];
#pragma unroll
    for (int r = 0; r < 8; ++r) {
      float x = fmaxf(s[0][r], s[1][r]);
      for (int off = 1; off < 16; off <<= 1) x = fmaxf(x, __shfl_xor(x, off, 32));
      float mnew = fmaxf(m_run[r], x);
      alpha[r] = __expf(m_run[r] - mnew);
      float p0 = __expf(s[0][r] - mnew);
      float p1 = __expf(s[1][r] - mnew);
      s[0][r] = p0; s[1][r] = p1;
      float rs = p0 + p1;
      for (int off = 1; off < 16; off <<= 1) rs += __shfl_xor(rs, off, 32);
      l_run[r] = alpha[r] * l_run[r] + rs;
      m_run[r] = mnew;
    }
#pragma unroll
    for (int t6 = 0; t6 < 6; ++t6)
#pragma unroll
      for (int r = 0; r < 8; ++r) o_acc[t6][r] *= alpha[r];

    // P (C layout) -> LDS in A-fragment order -> v16u load
    __syncthreads();
    {
      int rowoff = (lane < 16) ? 0 : 8;
#pragma unroll
      for (int t = 0; t < 2; ++t) {
        int sl = kslot(t * 16 + ml);
#pragma unroll
        for (int r = 0; r < 8; ++r)
          ps[(rowoff + r) * 32 + sl] = f2bf(s[t][r]);
      }
    }
    __syncthreads();

    v16u pa = *(const v16u*)&ps[ml * 32 + half * 16];
#pragma unroll
    for (int t6 = 0; t6 < 6; ++t6) {
      v16u vb = *(const v16u*)&vst[(t6 * 16 + ml) * 32 + half * 16];
      o_acc[t6] = wmma_bf16(pa, vb, o_acc[t6]);
    }
    __syncthreads();
  }

#pragma unroll
  for (int t6 = 0; t6 < 6; ++t6)
#pragma unroll
    for (int r = 0; r < 8; ++r) {
      int m = (lane < 16) ? r : 8 + r;
      int srow = qt * 16 + m;
      out[((long)(b * SEQ + srow)) * D_MODEL + h * HEAD_DIM + t6 * 16 + ml] =
          o_acc[t6][r] / l_run[r];
    }
}

// ---------------------------------------------------------------------------
// Concatenate Wq|Wk|Wv -> Wqkv [768 x 2304] and biases -> bqkv [2304]
// ---------------------------------------------------------------------------
__global__ void concat_qkv_kernel(const float* __restrict__ Wq,
                                  const float* __restrict__ Wk,
                                  const float* __restrict__ Wv,
                                  const float* __restrict__ bq,
                                  const float* __restrict__ bk,
                                  const float* __restrict__ bv,
                                  float* __restrict__ Wqkv,
                                  float* __restrict__ bqkv) {
  long i = (long)blockIdx.x * 256 + threadIdx.x;
  const long total = (long)D_MODEL * D_MODEL;
  if (i < total) {
    long kk = i / D_MODEL, n = i % D_MODEL;
    long o = kk * QKV_N;
    Wqkv[o + n]               = Wq[i];
    Wqkv[o + D_MODEL + n]     = Wk[i];
    Wqkv[o + 2 * D_MODEL + n] = Wv[i];
  }
  if (i < D_MODEL) {
    bqkv[i]               = bq[i];
    bqkv[D_MODEL + i]     = bk[i];
    bqkv[2 * D_MODEL + i] = bv[i];
  }
}

// ---------------------------------------------------------------------------
// src = concat(prefix, hand + mod, head + mod); coords = concat(0, ch, che)
// ---------------------------------------------------------------------------
__global__ void build_src_kernel(const float* __restrict__ hand,
                                 const float* __restrict__ headt,
                                 const float* __restrict__ prefix,
                                 const float* __restrict__ mod,
                                 float* __restrict__ src) {
  long i = (long)blockIdx.x * 256 + threadIdx.x;
  const long total = (long)ROWS * D_MODEL;
  if (i >= total) return;
  int d = (int)(i % D_MODEL);
  long bs = i / D_MODEL;
  int s = (int)(bs % SEQ), b = (int)(bs / SEQ);
  float vv;
  if (s < PREFIX_N)       vv = prefix[s * D_MODEL + d];
  else if (s < PREFIX_N + NTOK)
    vv = hand[((long)(b * NTOK + (s - PREFIX_N))) * D_MODEL + d] + mod[d];
  else
    vv = headt[((long)(b * NTOK + (s - PREFIX_N - NTOK))) * D_MODEL + d] + mod[d];
  src[i] = vv;
}

__global__ void build_coords_kernel(const float* __restrict__ ch,
                                    const float* __restrict__ che,
                                    float* __restrict__ coords) {
  long i = (long)blockIdx.x * 256 + threadIdx.x;
  const long total = (long)ROWS * 3;
  if (i >= total) return;
  int d = (int)(i % 3);
  long bs = i / 3;
  int s = (int)(bs % SEQ), b = (int)(bs / SEQ);
  float vv;
  if (s < PREFIX_N)       vv = 0.f;
  else if (s < PREFIX_N + NTOK) vv = ch[((long)(b * NTOK + (s - PREFIX_N))) * 3 + d];
  else                          vv = che[((long)(b * NTOK + (s - PREFIX_N - NTOK))) * 3 + d];
  coords[i] = vv;
}

// ---------------------------------------------------------------------------
// 3-axis RoPE on q and k (views into qkv, row stride ld) in place.
// dim index within head = blk*6 + axis*2 (+0/+1); theta = 10000^(-blk/16).
// ---------------------------------------------------------------------------
__global__ void rope_kernel(float* __restrict__ q, float* __restrict__ k,
                            const float* __restrict__ coords, long ld) {
  long i = (long)blockIdx.x * 256 + threadIdx.x;
  const long total = (long)BATCH * SEQ * N_HEADS * 48;
  if (i >= total) return;
  int pair = (int)(i % 48);
  long t = i / 48;
  int h = (int)(t % N_HEADS); t /= N_HEADS;
  int s = (int)(t % SEQ);
  int b = (int)(t / SEQ);
  int axis = pair % 3, blk = pair / 3;
  float theta = __powf(10000.0f, -(float)blk / 16.0f);
  float ang = coords[((long)(b * SEQ + s)) * 3 + axis] * theta;
  float c = __cosf(ang), sn = __sinf(ang);
  long base = ((long)(b * SEQ + s)) * ld + h * HEAD_DIM + blk * 6 + axis * 2;
  float e = q[base], o = q[base + 1];
  q[base] = e * c - o * sn;  q[base + 1] = e * sn + o * c;
  e = k[base];  o = k[base + 1];
  k[base] = e * c - o * sn;  k[base + 1] = e * sn + o * c;
}

// ---------------------------------------------------------------------------
// out = LN(x + res) * g + be  [optional relu]. One 256-thread block per row.
// wave32 shuffle reduction + single cross-wave stage (one barrier).
// ---------------------------------------------------------------------------
__global__ __launch_bounds__(256) void ln_kernel(
    const float* __restrict__ x, const float* __restrict__ res,
    const float* __restrict__ g, const float* __restrict__ be,
    float* __restrict__ out, int rowlen, int relu) {
  const int row = blockIdx.x, tid = threadIdx.x;
  const int lane = tid & 31, wave = tid >> 5;
  const float* xr = x + (long)row * rowlen;
  const float* rr = res ? res + (long)row * rowlen : nullptr;
  float s = 0.f, ss = 0.f;
  for (int i = tid; i < rowlen; i += 256) {
    float t = xr[i] + (rr ? rr[i] : 0.f);
    s += t; ss += t * t;
  }
#pragma unroll
  for (int off = 1; off < 32; off <<= 1) {
    s  += __shfl_xor(s,  off, 32);
    ss += __shfl_xor(ss, off, 32);
  }
  __shared__ float sm[8], sv[8];
  if (lane == 0) { sm[wave] = s; sv[wave] = ss; }
  __syncthreads();
  float tot = 0.f, tot2 = 0.f;
#pragma unroll
  for (int w = 0; w < 8; ++w) { tot += sm[w]; tot2 += sv[w]; }
  float mean = tot / rowlen;
  float inv  = rsqrtf(tot2 / rowlen - mean * mean + 1e-5f);
  for (int i = tid; i < rowlen; i += 256) {
    float t = xr[i] + (rr ? rr[i] : 0.f);
    float y = (t - mean) * inv * g[i] + be[i];
    if (relu) y = fmaxf(y, 0.f);
    out[(long)row * rowlen + i] = y;
  }
}

// ---------------------------------------------------------------------------
// Host-side orchestration
// ---------------------------------------------------------------------------
static inline void launch_gemm(hipStream_t st, const float* A, long lda,
                               const float* W, const float* bias, float* C,
                               long ldc, int M, int N, int K, int act) {
  dim3 grid(N / 64, (M + 127) / 128);
  gemm_bf16_kernel<<<grid, 256, 0, st>>>(A, lda, W, bias, C, ldc, M, N, K, act);
}

extern "C" void kernel_launch(void* const* d_in, const int* in_sizes, int n_in,
                              void* d_out, int out_size, void* d_ws, size_t ws_size,
                              hipStream_t stream) {
  (void)in_sizes; (void)n_in; (void)out_size; (void)ws_size;
  const float* hand   = (const float*)d_in[0];
  const float* headt  = (const float*)d_in[1];
  const float* chand  = (const float*)d_in[2];
  const float* chead  = (const float*)d_in[3];
  const float* prefix = (const float*)d_in[4];
  const float* mod    = (const float*)d_in[5];
  const float* mW1 = (const float*)d_in[38];
  const float* mb1 = (const float*)d_in[39];
  const float* mg  = (const float*)d_in[40];
  const float* mbe = (const float*)d_in[41];
  const float* mW2 = (const float*)d_in[42];
  const float* mb2 = (const float*)d_in[43];

  float* ws = (float*)d_ws;
  const size_t nsd = (size_t)ROWS * D_MODEL;
  float* src    = ws; ws += nsd;
  float* coords = ws; ws += (size_t)ROWS * 3 + 64;
  float* qkv    = ws; ws += (size_t)ROWS * QKV_N;
  float* wqkv   = ws; ws += (size_t)D_MODEL * QKV_N;
  float* bqkv   = ws; ws += QKV_N;
  float* attn   = ws; ws += nsd;
  float* tmp    = ws; ws += nsd;
  float* x1     = ws; ws += nsd;
  float* hid    = ws; ws += (size_t)ROWS * FFN_DIM;
  float* hh     = ws; ws += (size_t)BATCH * MLP_HID_N;
  float* hh2    = ws; ws += (size_t)BATCH * MLP_HID_N;

  {
    long tot = (long)ROWS * D_MODEL;
    build_src_kernel<<<(unsigned)((tot + 255) / 256), 256, 0, stream>>>(hand, headt, prefix, mod, src);
    long tc = (long)ROWS * 3;
    build_coords_kernel<<<(unsigned)((tc + 255) / 256), 256, 0, stream>>>(chand, chead, coords);
  }

  for (int l = 0; l < 2; ++l) {
    int base = 6 + l * 16;
    const float* Wq = (const float*)d_in[base + 0];
    const float* bq = (const float*)d_in[base + 1];
    const float* Wk = (const float*)d_in[base + 2];
    const float* bk = (const float*)d_in[base + 3];
    const float* Wv = (const float*)d_in[base + 4];
    const float* bv = (const float*)d_in[base + 5];
    const float* Wo = (const float*)d_in[base + 6];
    const float* bo = (const float*)d_in[base + 7];
    const float* W1 = (const float*)d_in[base + 8];
    const float* b1 = (const float*)d_in[base + 9];
    const float* W2 = (const float*)d_in[base + 10];
    const float* b2 = (const float*)d_in[base + 11];
    const float* g1 = (const float*)d_in[base + 12];
    const float* be1= (const float*)d_in[base + 13];
    const float* g2 = (const float*)d_in[base + 14];
    const float* be2= (const float*)d_in[base + 15];

    // fused QKV: concat weights once, one pass over src
    {
      long tot = (long)D_MODEL * D_MODEL;
      concat_qkv_kernel<<<(unsigned)((tot + 255) / 256), 256, 0, stream>>>(
          Wq, Wk, Wv, bq, bk, bv, wqkv, bqkv);
    }
    launch_gemm(stream, src, D_MODEL, wqkv, bqkv, qkv, QKV_N, ROWS, QKV_N, D_MODEL, 0);

    {
      long tot = (long)BATCH * SEQ * N_HEADS * 48;
      rope_kernel<<<(unsigned)((tot + 255) / 256), 256, 0, stream>>>(
          qkv, qkv + D_MODEL, coords, QKV_N);
    }

    attn_kernel<<<BATCH * N_HEADS * (SEQ / 16), 32, 0, stream>>>(
        qkv, qkv + D_MODEL, qkv + 2 * D_MODEL, attn, QKV_N);

    launch_gemm(stream, attn, D_MODEL, Wo, bo, tmp, D_MODEL, ROWS, D_MODEL, D_MODEL, 0);
    ln_kernel<<<ROWS, 256, 0, stream>>>(tmp, src, g1, be1, x1, D_MODEL, 0);

    launch_gemm(stream, x1, D_MODEL, W1, b1, hid, FFN_DIM, ROWS, FFN_DIM, D_MODEL, 1);  // GELU
    launch_gemm(stream, hid, FFN_DIM, W2, b2, tmp, D_MODEL, ROWS, D_MODEL, FFN_DIM, 0);
    ln_kernel<<<ROWS, 256, 0, stream>>>(tmp, x1, g2, be2, src, D_MODEL, 0);
  }

  // Head MLP: prefix rows of src are contiguous per batch -> A with lda = SEQ*D
  launch_gemm(stream, src, (long)SEQ * D_MODEL, mW1, mb1, hh, MLP_HID_N,
              BATCH, MLP_HID_N, PREFIX_N * D_MODEL, 0);
  ln_kernel<<<BATCH, 256, 0, stream>>>(hh, nullptr, mg, mbe, hh2, MLP_HID_N, 1); // LN + ReLU
  launch_gemm(stream, hh2, MLP_HID_N, mW2, mb2, (float*)d_out, OUT_DIM_N,
              BATCH, OUT_DIM_N, MLP_HID_N, 0);
}
